// MultiHeadAttention_73847667688118
// MI455X (gfx1250) — matrix operand order repaired and checked by reading, not compiled, not run on hardware
//
#include <hip/hip_runtime.h>
#include <hip/hip_bf16.h>
#include <stdint.h>

// Problem constants (from reference)
#define B_  4
#define S_  2048
#define D_  1024
#define H_  16
#define HD_ 64
#define M_  (B_ * S_)   // 8192 rows for projection GEMMs

typedef __attribute__((ext_vector_type(16))) __bf16 v16bf;
typedef __attribute__((ext_vector_type(8)))  float  v8f;

union Frag16 { v16bf v; uint4 q[2]; unsigned short s[16]; };
union FragC  { v8f v; float f[8]; };

__device__ __forceinline__ unsigned short f2bf(float x) {
  union { float f; unsigned u; } c; c.f = x;
  unsigned u = c.u;
  u += 0x7FFFu + ((u >> 16) & 1u);   // round-to-nearest-even
  return (unsigned short)(u >> 16);
}

// ---------------------------------------------------------------------------
// Tensor Data Mover: 2-D bf16 tile -> LDS with hardware row padding.
// Descriptor layout per CDNA5 ISA §8.3 (group0) / §8.4 (group1).
// ---------------------------------------------------------------------------
#if __has_builtin(__builtin_amdgcn_tensor_load_to_lds)
#define HAVE_TDM 1
typedef unsigned int u32x4 __attribute__((ext_vector_type(4)));
typedef int          i32x4 __attribute__((ext_vector_type(4)));
typedef int          i32x8 __attribute__((ext_vector_type(8)));

__device__ __forceinline__ void tdm_load_2d_bf16(
    const void* gsrc, unsigned lds_addr,
    unsigned tensor_d0, unsigned tensor_d1,   // tensor extents (elements)
    unsigned tile_d0, unsigned tile_d1,       // tile extents (elements)
    unsigned stride0,                         // dim0 row stride (elements)
    unsigned pad_interval_code,               // 1<<(c+1) DWORDs between pads
    unsigned pad_amount_code)                 // (c+1) DWORDs of pad
{
  const unsigned long long ga = (unsigned long long)(uintptr_t)gsrc;
  u32x4 g0;
  g0.x = 1u;                                             // count=1, user desc
  g0.y = lds_addr;                                       // LDS byte address
  g0.z = (unsigned)(ga & 0xffffffffu);                   // global_addr[31:0]
  g0.w = (unsigned)((ga >> 32) & 0x1ffffffu) | (2u << 30); // ga[56:32] | type=2
  i32x8 g1;
  g1[0] = (int)((1u << 16)                               // data_size = 2 bytes
              | (1u << 20)                               // pad_enable
              | (pad_interval_code << 22)
              | (pad_amount_code << 25));
  g1[1] = (int)((tensor_d0 & 0xffffu) << 16);            // ab_addr=0 | dim0 lo
  g1[2] = (int)(((tensor_d0 >> 16) & 0xffffu)
              | ((tensor_d1 & 0xffffu) << 16));          // dim0 hi | dim1 lo
  g1[3] = (int)(((tensor_d1 >> 16) & 0xffffu)
              | ((tile_d0 & 0xffffu) << 16));            // dim1 hi | tile_dim0
  g1[4] = (int)(tile_d1 & 0xffffu);                      // tile_dim1 | tile_dim2=0
  g1[5] = (int)stride0;                                  // dim0_stride[31:0]
  g1[6] = 0;                                             // stride hi bits = 0
  g1[7] = 0;
  const i32x4 z4 = {0, 0, 0, 0};
#if __clang_major__ >= 23
  const i32x8 z8 = {0, 0, 0, 0, 0, 0, 0, 0};
  __builtin_amdgcn_tensor_load_to_lds(g0, g1, z4, z4, z8, 0);
#else
  __builtin_amdgcn_tensor_load_to_lds(g0, g1, z4, z4, 0);
#endif
}
#else
#define HAVE_TDM 0
#endif

// ---------------------------------------------------------------------------
// GEMM: C[M_,1024] = A[M_,1024] @ W[1024,1024]
//   A_BF16  : 0 -> A is fp32, 1 -> A is bf16 (TDM-staged when available)
//   OUT_HEAD: 1 -> write bf16 in head layout [B,H,S,HD]; 0 -> fp32 row-major
// Block: 256 threads (8 waves). Tile 128x128, K-step 32.
// ---------------------------------------------------------------------------
template <int A_BF16, int OUT_HEAD>
__global__ __launch_bounds__(256)
void gemm128(const void* __restrict__ Av, const float* __restrict__ W,
             void* __restrict__ Outv) {
  __shared__ __align__(16) unsigned short As[128][40];  // [m][k], stride 80B = 64B data + 16B pad
  __shared__ __align__(16) unsigned short Bt[128][40];  // [n][k], transposed W tile

  const int tid  = threadIdx.x;
  const int wave = tid >> 5, lane = tid & 31;
  const int lm = lane & 15, hh = lane >> 4;
  const int h8 = hh * 8, h16 = hh * 16;
  const int wm = (wave & 3) * 32;
  const int wn = (wave >> 2) * 64;
  const int bm = blockIdx.y * 128;
  const int bn = blockIdx.x * 128;

  const float*          Af = (const float*)Av;
  const unsigned short* Ab = (const unsigned short*)Av;

  FragC acc[2][4];
#pragma unroll
  for (int i = 0; i < 2; ++i)
#pragma unroll
    for (int j = 0; j < 4; ++j)
#pragma unroll
      for (int r = 0; r < 8; ++r) acc[i][j].f[r] = 0.0f;

  const int am = tid >> 1, akh = (tid & 1) * 16;     // A-tile loader mapping
  const int wk = tid >> 3, wnc = (tid & 7) * 16;     // W-tile loader mapping

  for (int k0 = 0; k0 < D_; k0 += 32) {
    __syncthreads();
    // ---- stage A tile (128 x 32) as bf16 ----
#if HAVE_TDM
    if (A_BF16) {
      if (wave == 0) {
        // 128 rows x 32 bf16 (64B) + 16B pad/row -> As stride 80B.
        // pad_interval_code 3 -> 16 DWORDs (64B); pad_amount_code 3 -> 4 DWORDs (16B).
        tdm_load_2d_bf16(&Ab[(size_t)bm * D_ + k0],
                         (unsigned)(uintptr_t)&As[0][0],
                         /*tensor_d0=*/D_, /*tensor_d1=*/M_ - bm,
                         /*tile_d0=*/32, /*tile_d1=*/128,
                         /*stride0=*/D_, 3u, 3u);
      }
    } else
#endif
    if (A_BF16) {
      const uint4* src = (const uint4*)&Ab[(size_t)(bm + am) * D_ + k0 + akh];
      *(uint4*)&As[am][akh]     = src[0];
      *(uint4*)&As[am][akh + 8] = src[1];
      if (k0 + 32 < D_) __builtin_prefetch(&Ab[(size_t)(bm + am) * D_ + k0 + 32 + akh], 0, 3);
    } else {
      const float4* src = (const float4*)&Af[(size_t)(bm + am) * D_ + k0 + akh];
      unsigned short tmp[16];
#pragma unroll
      for (int j = 0; j < 4; ++j) {
        float4 v = src[j];
        tmp[4 * j + 0] = f2bf(v.x); tmp[4 * j + 1] = f2bf(v.y);
        tmp[4 * j + 2] = f2bf(v.z); tmp[4 * j + 3] = f2bf(v.w);
      }
#pragma unroll
      for (int j = 0; j < 16; ++j) As[am][akh + j] = tmp[j];
      if (k0 + 32 < D_) __builtin_prefetch(&Af[(size_t)(bm + am) * D_ + k0 + 32 + akh], 0, 3);
    }
    // ---- stage W tile (32 x 128) transposed to [n][k] as bf16 ----
    {
      const float4* src = (const float4*)&W[(size_t)(k0 + wk) * D_ + bn + wnc];
#pragma unroll
      for (int j = 0; j < 4; ++j) {
        float4 v = src[j];
        Bt[wnc + 4 * j + 0][wk] = f2bf(v.x);
        Bt[wnc + 4 * j + 1][wk] = f2bf(v.y);
        Bt[wnc + 4 * j + 2][wk] = f2bf(v.z);
        Bt[wnc + 4 * j + 3][wk] = f2bf(v.w);
      }
      if (k0 + 32 < D_) __builtin_prefetch(&W[(size_t)(k0 + 32 + wk) * D_ + bn + wnc], 0, 3);
    }
#if HAVE_TDM
    if (A_BF16 && wave == 0) __builtin_amdgcn_s_wait_tensorcnt(0);
#endif
    __syncthreads();

    // ---- gather fragments (documented 16-bit A/B WMMA layouts) ----
    Frag16 af[2];
#pragma unroll
    for (int mt = 0; mt < 2; ++mt) {
      const int row = wm + mt * 16 + lm;
      af[mt].q[0] = *(const uint4*)&As[row][h8];        // K = h8..h8+7
      af[mt].q[1] = *(const uint4*)&As[row][16 + h8];   // K = 16+h8..
    }
    Frag16 bfg[4];
#pragma unroll
    for (int nt = 0; nt < 4; ++nt) {
      const int col = wn + nt * 16 + lm;
      bfg[nt].q[0] = *(const uint4*)&Bt[col][h8];       // K = h8..h8+7
      bfg[nt].q[1] = *(const uint4*)&Bt[col][16 + h8];  // K = 16+h8..
    }
#pragma unroll
    for (int mt = 0; mt < 2; ++mt)
#pragma unroll
      for (int nt = 0; nt < 4; ++nt)
        acc[mt][nt].v = __builtin_amdgcn_wmma_f32_16x16x32_bf16(
            false, af[mt].v, false, bfg[nt].v, (short)0, acc[mt][nt].v, false, false);
  }

  // ---- epilogue ----
  if (OUT_HEAD) {
    unsigned short* Ob = (unsigned short*)Outv;
#pragma unroll
    for (int mt = 0; mt < 2; ++mt)
#pragma unroll
      for (int nt = 0; nt < 4; ++nt)
#pragma unroll
        for (int r = 0; r < 8; ++r) {
          const int m = bm + wm + mt * 16 + r + 8 * hh;
          const int n = bn + wn + nt * 16 + lm;
          const int b = m >> 11, s = m & (S_ - 1);
          const int h = n >> 6,  d = n & (HD_ - 1);
          Ob[(((size_t)b * H_ + h) * S_ + s) * HD_ + d] = f2bf(acc[mt][nt].f[r]);
        }
  } else {
    float* Of = (float*)Outv;
#pragma unroll
    for (int mt = 0; mt < 2; ++mt)
#pragma unroll
      for (int nt = 0; nt < 4; ++nt)
#pragma unroll
        for (int r = 0; r < 8; ++r) {
          const int m = bm + wm + mt * 16 + r + 8 * hh;
          const int n = bn + wn + nt * 16 + lm;
          Of[(size_t)m * D_ + n] = acc[mt][nt].f[r];
        }
  }
}

// ---------------------------------------------------------------------------
// Flash attention: per (b,h), online softmax over 2048 keys, 32 keys/step.
// Block: 128 threads (4 waves); each wave owns a 16-row query tile and shares
// the K/V LDS tiles. K tile is TDM-staged (with HW row padding) when available.
// ---------------------------------------------------------------------------
__global__ __launch_bounds__(128)
void attn_fa(const unsigned short* __restrict__ Qh,
             const unsigned short* __restrict__ Kh,
             const unsigned short* __restrict__ Vh,
             const int* __restrict__ mask,
             unsigned short* __restrict__ O) {
  __shared__ __align__(16) unsigned short Ks[32][72];      // [key][d], stride 144B = 128B data + 16B pad
  __shared__ __align__(16) unsigned short Vt[64][40];      // [d][key], transposed
  __shared__ __align__(16) unsigned short Ps[4][16][40];   // per-wave P tile [qrow][key]

  const int tid  = threadIdx.x;
  const int wave = tid >> 5, lane = tid & 31;
  const int lm = lane & 15, hh = lane >> 4;
  const int h8 = hh * 8, h16 = hh * 16;
  const int b = blockIdx.z, h = blockIdx.y;
  const int bh = b * H_ + h;
  const int qbase = blockIdx.x * 64 + wave * 16;

  // Q fragments held in registers for the whole key loop
  Frag16 qf[2];
  {
    const unsigned short* qrow = &Qh[((size_t)bh * S_ + qbase + lm) * HD_];
#pragma unroll
    for (int c = 0; c < 2; ++c) {
      qf[c].q[0] = *(const uint4*)&qrow[c * 32 + h8];
      qf[c].q[1] = *(const uint4*)&qrow[c * 32 + 16 + h8];
    }
  }

  FragC o[4];
#pragma unroll
  for (int nt = 0; nt < 4; ++nt)
#pragma unroll
    for (int r = 0; r < 8; ++r) o[nt].f[r] = 0.0f;
  float rm[8], rs[8];
#pragma unroll
  for (int r = 0; r < 8; ++r) { rm[r] = -__builtin_inff(); rs[r] = 0.0f; }

  const int key0 = tid >> 2;        // 0..31 : cooperative loader key index
  const int dc   = (tid & 3) * 16;  // 0,16,32,48

  for (int kb = 0; kb < S_; kb += 32) {
    __syncthreads();
#if HAVE_TDM
    if (wave == 0) {
      // 32 keys x 64 bf16 (128B) + 16B pad/row -> Ks stride 144B.
      // pad_interval_code 4 -> 32 DWORDs (128B); pad_amount_code 3 -> 4 DWORDs (16B).
      tdm_load_2d_bf16(&Kh[((size_t)bh * S_ + kb) * HD_],
                       (unsigned)(uintptr_t)&Ks[0][0],
                       /*tensor_d0=*/HD_, /*tensor_d1=*/S_ - kb,
                       /*tile_d0=*/HD_, /*tile_d1=*/32,
                       /*stride0=*/HD_, 4u, 3u);
    }
#else
    {  // K tile: direct copy [key][d]
      const uint4* src = (const uint4*)&Kh[((size_t)bh * S_ + kb + key0) * HD_ + dc];
      *(uint4*)&Ks[key0][dc]     = src[0];
      *(uint4*)&Ks[key0][dc + 8] = src[1];
    }
#endif
    {  // V tile: transpose to [d][key]
      const uint4* src = (const uint4*)&Vh[((size_t)bh * S_ + kb + key0) * HD_ + dc];
      Frag16 tmp; tmp.q[0] = src[0]; tmp.q[1] = src[1];
#pragma unroll
      for (int j = 0; j < 16; ++j) Vt[dc + j][key0] = tmp.s[j];
    }
    if (kb + 32 < S_) {
      __builtin_prefetch(&Kh[((size_t)bh * S_ + kb + 32 + key0) * HD_ + dc], 0, 3);
      __builtin_prefetch(&Vh[((size_t)bh * S_ + kb + 32 + key0) * HD_ + dc], 0, 3);
    }
#if HAVE_TDM
    if (wave == 0) __builtin_amdgcn_s_wait_tensorcnt(0);
#endif
    __syncthreads();

    // ---- scores: S = Q(16x64) @ K^T(64x32) -> two 16x16 f32 tiles ----
    FragC sc[2];
#pragma unroll
    for (int t = 0; t < 2; ++t) {
#pragma unroll
      for (int r = 0; r < 8; ++r) sc[t].f[r] = 0.0f;
#pragma unroll
      for (int c = 0; c < 2; ++c) {
        Frag16 kf;
        kf.q[0] = *(const uint4*)&Ks[t * 16 + lm][c * 32 + h8];
        kf.q[1] = *(const uint4*)&Ks[t * 16 + lm][c * 32 + 16 + h8];
        sc[t].v = __builtin_amdgcn_wmma_f32_16x16x32_bf16(
            false, qf[c].v, false, kf.v, (short)0, sc[t].v, false, false);
      }
    }

    // ---- scale, mask, online softmax update ----
    const int keyA = kb + lm, keyB = kb + 16 + lm;
#pragma unroll
    for (int r = 0; r < 8; ++r) {
      const int qrow = qbase + r + 8 * hh;
      const int* mrow = &mask[((size_t)b * S_ + qrow) * S_];
      float x0 = sc[0].f[r] * 0.125f;   // 1/sqrt(64)
      float x1 = sc[1].f[r] * 0.125f;
      x0 = (mrow[keyA] == 0) ? -1e9f : x0;
      x1 = (mrow[keyB] == 0) ? -1e9f : x1;
      float tmax = fmaxf(x0, x1);
      tmax = fmaxf(tmax, __shfl_xor(tmax, 1, 32));
      tmax = fmaxf(tmax, __shfl_xor(tmax, 2, 32));
      tmax = fmaxf(tmax, __shfl_xor(tmax, 4, 32));
      tmax = fmaxf(tmax, __shfl_xor(tmax, 8, 32));   // stays within 16-lane half
      const float newm = fmaxf(rm[r], tmax);
      const float corr = __expf(rm[r] - newm);
      const float e0 = __expf(x0 - newm);
      const float e1 = __expf(x1 - newm);
      float ps = e0 + e1;
      ps += __shfl_xor(ps, 1, 32);
      ps += __shfl_xor(ps, 2, 32);
      ps += __shfl_xor(ps, 4, 32);
      ps += __shfl_xor(ps, 8, 32);
      rs[r] = rs[r] * corr + ps;
      rm[r] = newm;
      sc[0].f[r] = e0; sc[1].f[r] = e1;
#pragma unroll
      for (int nt = 0; nt < 4; ++nt) o[nt].f[r] *= corr;
    }

    // ---- P (C-layout) -> per-wave LDS -> A-layout fragment ----
#pragma unroll
    for (int r = 0; r < 8; ++r) {
      Ps[wave][r + 8 * hh][lm]      = f2bf(sc[0].f[r]);
      Ps[wave][r + 8 * hh][16 + lm] = f2bf(sc[1].f[r]);
    }
    asm volatile("s_wait_dscnt 0" ::: "memory");  // wave-internal LDS RAW
    Frag16 pf;
    pf.q[0] = *(const uint4*)&Ps[wave][lm][h8];
    pf.q[1] = *(const uint4*)&Ps[wave][lm][16 + h8];

    // ---- O += P(16x32) @ V(32x64) ----
#pragma unroll
    for (int nt = 0; nt < 4; ++nt) {
      Frag16 vf;
      vf.q[0] = *(const uint4*)&Vt[nt * 16 + lm][h8];
      vf.q[1] = *(const uint4*)&Vt[nt * 16 + lm][16 + h8];
      o[nt].v = __builtin_amdgcn_wmma_f32_16x16x32_bf16(
          false, pf.v, false, vf.v, (short)0, o[nt].v, false, false);
    }
  }

  // ---- normalize and write merged-head bf16 output [B,S,D] ----
#pragma unroll
  for (int nt = 0; nt < 4; ++nt)
#pragma unroll
    for (int r = 0; r < 8; ++r) {
      const int s = qbase + r + 8 * hh;
      const int d = nt * 16 + lm;
      O[((size_t)b * S_ + s) * D_ + h * HD_ + d] = f2bf(o[nt].f[r] / rs[r]);
    }
}

// ---------------------------------------------------------------------------
extern "C" void kernel_launch(void* const* d_in, const int* in_sizes, int n_in,
                              void* d_out, int out_size, void* d_ws, size_t ws_size,
                              hipStream_t stream) {
  (void)in_sizes; (void)n_in; (void)out_size; (void)ws_size;
  const float* q    = (const float*)d_in[0];
  const float* k    = (const float*)d_in[1];
  const float* v    = (const float*)d_in[2];
  const int*   mask = (const int*)d_in[3];
  const float* Wq   = (const float*)d_in[4];
  const float* Wk   = (const float*)d_in[5];
  const float* Wv   = (const float*)d_in[6];
  const float* Wo   = (const float*)d_in[7];

  // Workspace: 4 bf16 tensors of B*S*D = 16 MiB each (64 MiB total)
  const size_t SZ = (size_t)M_ * D_;
  unsigned short* Qh = (unsigned short*)d_ws;
  unsigned short* Kh = Qh + SZ;
  unsigned short* Vh = Kh + SZ;
  unsigned short* AO = Vh + SZ;

  const dim3 gGrid(D_ / 128, M_ / 128);  // 8 x 64 tiles
  gemm128<0, 1><<<gGrid, 256, 0, stream>>>(q, Wq, Qh);
  gemm128<0, 1><<<gGrid, 256, 0, stream>>>(k, Wk, Kh);
  gemm128<0, 1><<<gGrid, 256, 0, stream>>>(v, Wv, Vh);

  attn_fa<<<dim3(S_ / 64, H_, B_), 128, 0, stream>>>(Qh, Kh, Vh, mask, AO);

  gemm128<1, 0><<<gGrid, 256, 0, stream>>>(AO, Wo, (float*)d_out);
}
